// VQVAE_84516366450999
// MI455X (gfx1250) — matrix operand order repaired and checked by reading, not compiled, not run on hardware
//
#include <hip/hip_runtime.h>
#include <hip/hip_bf16.h>

typedef _Float16 f16;
typedef __attribute__((ext_vector_type(16))) _Float16 v16h;
typedef __attribute__((ext_vector_type(8)))  _Float16 v8h;
typedef __attribute__((ext_vector_type(8)))  float    v8f;

// Problem dims (match reference)
#define KCODES 1024
#define DDIM   64
#define NB     16
#define NVEC   65536          // 16 * 64 * 64 latent vectors
#define RECON_ELEMS (16*256*256)

// ---------------------------------------------------------------------------
// emb prep: f32 [1024,64] -> f16 rows + per-code squared norms
// ---------------------------------------------------------------------------
__global__ __launch_bounds__(256) void k_embprep(const float* __restrict__ emb,
                                                 f16* __restrict__ embh,
                                                 float* __restrict__ enorm) {
    int c = blockIdx.x * 256 + threadIdx.x;
    if (c >= KCODES) return;
    const float* e = emb + (size_t)c * DDIM;
    float s = 0.f;
    for (int d = 0; d < DDIM; ++d) {
        float v = e[d];
        s = fmaf(v, v, s);
        embh[(size_t)c * DDIM + d] = (f16)v;
    }
    enorm[c] = s;
}

// ---------------------------------------------------------------------------
// Encoder conv1: x[16,1,256,256] --(16ch,4x4,s2,p1)--> h1[16,16,128,128], relu
// ---------------------------------------------------------------------------
__global__ __launch_bounds__(256) void k_conv1(const float* __restrict__ x,
                                               const float* __restrict__ w,
                                               const float* __restrict__ bias,
                                               float* __restrict__ out) {
    int i  = blockIdx.x * 256 + threadIdx.x;      // 16*16*128*128
    int xo = i & 127, yo = (i >> 7) & 127, oc = (i >> 14) & 15, b = i >> 18;
    float acc = bias[oc];
    for (int ky = 0; ky < 4; ++ky) {
        int iy = yo * 2 - 1 + ky;
        if ((unsigned)iy >= 256u) continue;
        for (int kx = 0; kx < 4; ++kx) {
            int ix = xo * 2 - 1 + kx;
            if ((unsigned)ix >= 256u) continue;
            acc = fmaf(x[((size_t)b << 16) + iy * 256 + ix],
                       w[oc * 16 + ky * 4 + kx], acc);
        }
    }
    out[i] = fmaxf(acc, 0.f);
}

// ---------------------------------------------------------------------------
// Encoder conv2: h1 --(32ch,4x4,s2,p1)--> h2[16,32,64,64], relu
// ---------------------------------------------------------------------------
__global__ __launch_bounds__(256) void k_conv2(const float* __restrict__ h1,
                                               const float* __restrict__ w,
                                               const float* __restrict__ bias,
                                               float* __restrict__ out) {
    int i  = blockIdx.x * 256 + threadIdx.x;      // 16*32*64*64
    int xo = i & 63, yo = (i >> 6) & 63, oc = (i >> 12) & 31, b = i >> 17;
    float acc = bias[oc];
    for (int ic = 0; ic < 16; ++ic) {
        const float* hp = h1 + ((size_t)(b * 16 + ic) << 14);   // 128*128 plane
        const float* wp = w + ((oc * 16 + ic) << 4);
        for (int ky = 0; ky < 4; ++ky) {
            int iy = yo * 2 - 1 + ky;
            if ((unsigned)iy >= 128u) continue;
            for (int kx = 0; kx < 4; ++kx) {
                int ix = xo * 2 - 1 + kx;
                if ((unsigned)ix >= 128u) continue;
                acc = fmaf(hp[iy * 128 + ix], wp[ky * 4 + kx], acc);
            }
        }
    }
    out[i] = fmaxf(acc, 0.f);
}

// ---------------------------------------------------------------------------
// Encoder conv3 (1x1, 32->64) fused with VQ prep:
//   writes z as NHWC f16 rows [65536,64] + per-row f32 squared norm
// ---------------------------------------------------------------------------
__global__ __launch_bounds__(64) void k_conv3(const float* __restrict__ h2,
                                              const float* __restrict__ w,
                                              const float* __restrict__ bias,
                                              f16* __restrict__ zh,
                                              float* __restrict__ fnorm) {
    const int n  = blockIdx.x;              // 0..65535  == (b*64+h)*64+w
    const int b  = n >> 12;
    const int hw = n & 4095;
    const int d  = threadIdx.x;             // 0..63
    __shared__ float s_in[32];
    __shared__ float s_sq[64];
    if (d < 32) s_in[d] = h2[(((size_t)(b * 32 + d)) << 12) + hw];
    __syncthreads();
    float acc = bias[d];
    const float* wr = w + d * 32;           // e3_w is (64,32,1,1)
    for (int ic = 0; ic < 32; ++ic) acc = fmaf(s_in[ic], wr[ic], acc);
    zh[(size_t)n * DDIM + d] = (f16)acc;
    s_sq[d] = acc * acc;
    __syncthreads();
    if (d == 0) {
        float s = 0.f;
        for (int i = 0; i < 64; ++i) s += s_sq[i];
        fnorm[n] = s;
    }
}

// ---------------------------------------------------------------------------
// VQ nearest-code search via WMMA.
// Each wave owns 16 rows; loops 64 tiles of 16 codes; G = z @ emb^T via
// v_wmma_f32_16x16x32_f16 (K=64 as two k-chunks); d = |f|^2 + |e|^2 - 2G.
// Per-lane running argmin, LDS cross-lane reduce, then:
//   idx -> d_out (as float), q = emb[idx] (f32 NHWC) -> ws, min-d partial sums.
// ---------------------------------------------------------------------------
__global__ __launch_bounds__(256) void k_vq(const f16* __restrict__ zh,
                                            const float* __restrict__ fnorm,
                                            const f16* __restrict__ embh,
                                            const float* __restrict__ enorm,
                                            const float* __restrict__ emb32,
                                            float* __restrict__ q,
                                            float* __restrict__ idx_out,
                                            float* __restrict__ blocksum) {
    const int tid  = threadIdx.x;
    const int wave = tid >> 5;
    const int lane = tid & 31;
    const int hi   = lane >> 4;             // half-wave select
    const int ln   = lane & 15;
    const int rowbase = blockIdx.x * 128 + wave * 16;

    // A fragments: row (rowbase+ln) of zh, per ISA 16-bit A layout.
    // lane<16: k = j (j<8) / j+8 (j>=8); lane>=16: +8. Two chunks kbase=0,32.
    const f16* arow = zh + (size_t)(rowbase + ln) * DDIM;
    const int off = hi * 8;
    v16h a0, a1;
    {
        v8h p0 = *(const v8h*)(arow + off);
        v8h p1 = *(const v8h*)(arow + 16 + off);
        v8h p2 = *(const v8h*)(arow + 32 + off);
        v8h p3 = *(const v8h*)(arow + 48 + off);
        for (int i = 0; i < 8; ++i) {
            a0[i] = p0[i]; a0[i + 8] = p1[i];
            a1[i] = p2[i]; a1[i + 8] = p3[i];
        }
    }
    float fn[8];
    for (int r = 0; r < 8; ++r) fn[r] = fnorm[rowbase + r + hi * 8];

    float minv[8]; int mini[8];
    for (int r = 0; r < 8; ++r) { minv[r] = 3.4e38f; mini[r] = 0; }

    for (int t = 0; t < 64; ++t) {
        const int code = t * 16 + ln;
        const f16* brow = embh + (size_t)code * DDIM;
        if (t < 63)
            __builtin_prefetch(brow + DDIM * 16, 0, 0);   // global_prefetch_b8
        // B fragment: col = ln (code), k = j + hi*8 (j<8) / j+8+hi*8 (j>=8) within each k-chunk.
        v16h b0, b1;
        {
            v8h p0 = *(const v8h*)(brow + hi * 8);
            v8h p1 = *(const v8h*)(brow + hi * 8 + 16);
            v8h p2 = *(const v8h*)(brow + 32 + hi * 8);
            v8h p3 = *(const v8h*)(brow + 32 + hi * 8 + 16);
            for (int i = 0; i < 8; ++i) {
                b0[i] = p0[i]; b0[i + 8] = p1[i];
                b1[i] = p2[i]; b1[i + 8] = p3[i];
            }
        }
        v8f c = {};
        c = __builtin_amdgcn_wmma_f32_16x16x32_f16(false, a0, false, b0,
                                                   (short)0, c, false, false);
        c = __builtin_amdgcn_wmma_f32_16x16x32_f16(false, a1, false, b1,
                                                   (short)0, c, false, false);
        const float en = enorm[code];
        for (int r = 0; r < 8; ++r) {
            float d = fn[r] + en - 2.0f * c[r];
            if (d < minv[r]) { minv[r] = d; mini[r] = code; }
        }
    }

    // cross-lane reduce: row m's candidates live across the 16 columns
    __shared__ float s_v[8][16][16];
    __shared__ int   s_i[8][16][16];
    __shared__ float s_red[256];
    for (int r = 0; r < 8; ++r) {
        s_v[wave][r + hi * 8][ln] = minv[r];
        s_i[wave][r + hi * 8][ln] = mini[r];
    }
    __syncthreads();

    float contrib = 0.f;
    if (lane < 16) {
        const int m = lane;
        float bv = s_v[wave][m][0];
        int   bi = s_i[wave][m][0];
        for (int cN = 1; cN < 16; ++cN) {
            float v = s_v[wave][m][cN];
            if (v < bv) { bv = v; bi = s_i[wave][m][cN]; }
        }
        const int row = rowbase + m;
        idx_out[row] = (float)bi;
        const float* erow = emb32 + (size_t)bi * DDIM;   // q = gathered code
        float* qrow = q + (size_t)row * DDIM;
        for (int d = 0; d < DDIM; ++d) qrow[d] = erow[d];
        contrib = bv;                                    // = ||q - z||^2 (row)
    }
    s_red[tid] = contrib;
    __syncthreads();
    if (tid == 0) {
        float s = 0.f;
        for (int i = 0; i < 256; ++i) s += s_red[i];
        blocksum[blockIdx.x] = s;
    }
}

// deterministic fixed-order loss finish: loss = 1.25 * sum(min_d) / (N*D)
__global__ void k_finish(const float* __restrict__ blocksum,
                         float* __restrict__ loss) {
    if (blockIdx.x == 0 && threadIdx.x == 0) {
        float s = 0.f;
        for (int i = 0; i < 512; ++i) s += blocksum[i];
        *loss = 1.25f * s / (float)((size_t)NVEC * DDIM);
    }
}

// ---------------------------------------------------------------------------
// Decoder d1: q(NHWC f32) --(3x3, 64->32, p1)--> g1[16,32,64,64], relu
// ---------------------------------------------------------------------------
__global__ __launch_bounds__(256) void k_d1(const float* __restrict__ q,
                                            const float* __restrict__ w,
                                            const float* __restrict__ bias,
                                            float* __restrict__ out) {
    int i  = blockIdx.x * 256 + threadIdx.x;      // 16*32*64*64
    int xo = i & 63, yo = (i >> 6) & 63, oc = (i >> 12) & 31, b = i >> 17;
    float acc = bias[oc];
    for (int dy = 0; dy < 3; ++dy) {
        int iy = yo + dy - 1;
        if ((unsigned)iy >= 64u) continue;
        for (int dx = 0; dx < 3; ++dx) {
            int ix = xo + dx - 1;
            if ((unsigned)ix >= 64u) continue;
            const float* qrow = q + ((size_t)((b * 64 + iy) * 64 + ix)) * DDIM;
            const float* wp = w + oc * 576 + dy * 3 + dx;  // (32,64,3,3)
            for (int ic = 0; ic < 64; ++ic)
                acc = fmaf(qrow[ic], wp[ic * 9], acc);
        }
    }
    out[i] = fmaxf(acc, 0.f);
}

// ---------------------------------------------------------------------------
// Decoder d2: conv-transpose 32->16, k4, s2, p1 -> [16,16,128,128], relu
// ---------------------------------------------------------------------------
__global__ __launch_bounds__(256) void k_d2(const float* __restrict__ g1,
                                            const float* __restrict__ w,
                                            const float* __restrict__ bias,
                                            float* __restrict__ out) {
    int i  = blockIdx.x * 256 + threadIdx.x;      // 16*16*128*128
    int xo = i & 127, yo = (i >> 7) & 127, oc = (i >> 14) & 15, b = i >> 18;
    float acc = bias[oc];
    for (int ky = 0; ky < 4; ++ky) {
        int ty = yo + 1 - ky;
        if (ty & 1) continue;
        int iy = ty >> 1;
        if ((unsigned)iy >= 64u) continue;
        for (int kx = 0; kx < 4; ++kx) {
            int tx = xo + 1 - kx;
            if (tx & 1) continue;
            int ix = tx >> 1;
            if ((unsigned)ix >= 64u) continue;
            for (int ic = 0; ic < 32; ++ic)
                acc = fmaf(g1[(((size_t)(b * 32 + ic)) << 12) + iy * 64 + ix],
                           w[((ic * 16 + oc) * 4 + ky) * 4 + kx], acc);
        }
    }
    out[i] = fmaxf(acc, 0.f);
}

// ---------------------------------------------------------------------------
// Decoder d3: conv-transpose 16->1, k4, s2, p1 -> [16,1,256,256], sigmoid
// ---------------------------------------------------------------------------
__global__ __launch_bounds__(256) void k_d3(const float* __restrict__ g2,
                                            const float* __restrict__ w,
                                            const float* __restrict__ bias,
                                            float* __restrict__ recon) {
    int i  = blockIdx.x * 256 + threadIdx.x;      // 16*256*256
    int xo = i & 255, yo = (i >> 8) & 255, b = i >> 16;
    float acc = bias[0];
    for (int ky = 0; ky < 4; ++ky) {
        int ty = yo + 1 - ky;
        if (ty & 1) continue;
        int iy = ty >> 1;
        if ((unsigned)iy >= 128u) continue;
        for (int kx = 0; kx < 4; ++kx) {
            int tx = xo + 1 - kx;
            if (tx & 1) continue;
            int ix = tx >> 1;
            if ((unsigned)ix >= 128u) continue;
            for (int ic = 0; ic < 16; ++ic)
                acc = fmaf(g2[(((size_t)(b * 16 + ic)) << 14) + iy * 128 + ix],
                           w[ic * 16 + ky * 4 + kx], acc);   // (16,1,4,4)
        }
    }
    recon[i] = 1.0f / (1.0f + __expf(-acc));
}

// ---------------------------------------------------------------------------
extern "C" void kernel_launch(void* const* d_in, const int* in_sizes, int n_in,
                              void* d_out, int out_size, void* d_ws, size_t ws_size,
                              hipStream_t stream) {
    const float* x    = (const float*)d_in[0];
    const float* e1w  = (const float*)d_in[1];
    const float* e1b  = (const float*)d_in[2];
    const float* e2w  = (const float*)d_in[3];
    const float* e2b  = (const float*)d_in[4];
    const float* e3w  = (const float*)d_in[5];
    const float* e3b  = (const float*)d_in[6];
    const float* emb  = (const float*)d_in[7];
    const float* d1w  = (const float*)d_in[8];
    const float* d1b  = (const float*)d_in[9];
    const float* d2w  = (const float*)d_in[10];
    const float* d2b  = (const float*)d_in[11];
    const float* d3w  = (const float*)d_in[12];
    const float* d3b  = (const float*)d_in[13];

    float* out    = (float*)d_out;
    float* recon  = out;                      // 1,048,576 f32
    float* loss   = out + RECON_ELEMS;        // 1 f32
    float* idxf   = out + RECON_ELEMS + 1;    // 65,536 (idx as float)

    // workspace carve-up (256B aligned)
    char* ws = (char*)d_ws;
    size_t o = 0;
    auto carve = [&](size_t bytes) {
        void* p = ws + o;
        o = (o + bytes + 255) & ~(size_t)255;
        return p;
    };
    f16*   embh  = (f16*)  carve((size_t)KCODES * DDIM * 2);   // 128 KB
    float* enorm = (float*)carve((size_t)KCODES * 4);          //   4 KB
    float* fnorm = (float*)carve((size_t)NVEC * 4);            // 256 KB
    float* bsum  = (float*)carve(512 * 4);                     //   2 KB
    float* h1    = (float*)carve((size_t)16*16*128*128 * 4);   //  16 MB
    float* h2    = (float*)carve((size_t)16*32*64*64  * 4);    //   8 MB
    f16*   zh    = (f16*)  carve((size_t)NVEC * DDIM * 2);     //   8 MB
    float* q     = (float*)carve((size_t)NVEC * DDIM * 4);     //  16 MB
    float* g1    = (float*)carve((size_t)16*32*64*64  * 4);    //   8 MB
    float* g2    = (float*)carve((size_t)16*16*128*128 * 4);   //  16 MB
    (void)ws_size; (void)in_sizes; (void)n_in; (void)out_size;

    k_embprep<<<4,     256, 0, stream>>>(emb, embh, enorm);
    k_conv1  <<<16384, 256, 0, stream>>>(x, e1w, e1b, h1);
    k_conv2  <<<8192,  256, 0, stream>>>(h1, e2w, e2b, h2);
    k_conv3  <<<NVEC,  64,  0, stream>>>(h2, e3w, e3b, zh, fnorm);
    k_vq     <<<512,   256, 0, stream>>>(zh, fnorm, embh, enorm, emb, q, idxf, bsum);
    k_finish <<<1,     64,  0, stream>>>(bsum, loss);
    k_d1     <<<8192,  256, 0, stream>>>(q, d1w, d1b, g1);
    k_d2     <<<16384, 256, 0, stream>>>(g1, d2w, d2b, g2);
    k_d3     <<<4096,  256, 0, stream>>>(g2, d3w, d3b, recon);
}
